// RotaryCrossAttention_21406117003436
// MI455X (gfx1250) — matrix operand order repaired and checked by reading, not compiled, not run on hardware
//
#include <hip/hip_runtime.h>
#include <hip/hip_bf16.h>
#include <cstdint>

#define B_     16
#define NQ_    1024
#define T_     2048
#define C_     4
#define CDIM_  256
#define DIM_   512
#define DH_    128
#define INNER_ 128
#define KTOT_  (T_*C_)   /* 8192 */

typedef __attribute__((ext_vector_type(16))) __bf16 v16bf;
typedef __attribute__((ext_vector_type(8)))  __bf16 bf16x8;
typedef __attribute__((ext_vector_type(8)))  float  v8f;
typedef int i32x4 __attribute__((vector_size(16)));   // matches builtin param pointee

static __device__ __forceinline__ unsigned short f2bfu(float f) {
  unsigned int u = __float_as_uint(f);
  u += 0x7fffu + ((u >> 16) & 1u);            // round-to-nearest-even
  return (unsigned short)(u >> 16);
}
static __device__ __forceinline__ __bf16 f2bf(float f) {
  unsigned short h = f2bfu(f);
  __bf16 r; __builtin_memcpy(&r, &h, 2); return r;
}
static __device__ __forceinline__ bf16x8 ld8(const unsigned short* p) {
  return *reinterpret_cast<const bf16x8*>(p);   // 16B aligned by construction
}

// ---- CDNA5 async global->LDS path (guarded; falls back to register staging) ----
#if __has_builtin(__builtin_amdgcn_global_load_async_to_lds_b128) && \
    __has_builtin(__builtin_amdgcn_s_wait_asynccnt)
#define ASYNC_LDS 1
#else
#define ASYNC_LDS 0
#endif

static __device__ __forceinline__ void cp16_to_lds(const unsigned short* g, unsigned short* l) {
#if ASYNC_LDS
  void* gnc = (void*)g;                       // drop const
  __builtin_amdgcn_global_load_async_to_lds_b128(
      (__attribute__((address_space(1))) i32x4*)gnc,
      (__attribute__((address_space(3))) i32x4*)(void*)l, 0, 0);
#else
  *reinterpret_cast<bf16x8*>(l) = ld8(g);
#endif
}

// ---------------- f32 -> bf16 weight conversion ----------------
__global__ void k_convert(const float* __restrict__ x, unsigned short* __restrict__ y, int n) {
  const int i = blockIdx.x * blockDim.x + threadIdx.x;
  if (i < n) y[i] = f2bfu(x[i]);
}

// ---------------- layernorm(x_query) @ w_q^T (tiny: 134 MF) ----------------
__global__ void k_ln_q_proj(const float* __restrict__ xq,
                            const float* __restrict__ lnw, const float* __restrict__ lnb,
                            const float* __restrict__ w_q, float* __restrict__ q_proj) {
  __shared__ float xn[DIM_];
  __shared__ float red[256];
  const int row = blockIdx.x, tid = threadIdx.x;
  const float* x = xq + (size_t)row * DIM_;
  float s = 0.f;
  for (int i = tid; i < DIM_; i += 256) s += x[i];
  red[tid] = s; __syncthreads();
  for (int o = 128; o > 0; o >>= 1) { if (tid < o) red[tid] += red[tid + o]; __syncthreads(); }
  const float mu = red[0] * (1.f / DIM_); __syncthreads();
  float v = 0.f;
  for (int i = tid; i < DIM_; i += 256) { float d = x[i] - mu; v += d * d; }
  red[tid] = v; __syncthreads();
  for (int o = 128; o > 0; o >>= 1) { if (tid < o) red[tid] += red[tid + o]; __syncthreads(); }
  const float rstd = rsqrtf(red[0] * (1.f / DIM_) + 1e-5f); __syncthreads();
  for (int i = tid; i < DIM_; i += 256) xn[i] = (x[i] - mu) * rstd * lnw[i] + lnb[i];
  __syncthreads();
  if (tid < INNER_) {
    const float* wr = w_q + (size_t)tid * DIM_;
    float acc = 0.f;
    #pragma unroll 4
    for (int i = 0; i < DIM_; i++) acc += wr[i] * xn[i];
    q_proj[(size_t)row * INNER_ + tid] = acc;
  }
}

// ---------------- per-batch rotary on Q -> bf16 ----------------
__global__ void k_rot_q(const float* __restrict__ q_proj, const float* __restrict__ rot_q,
                        unsigned short* __restrict__ q_rot) {
  const int idx = blockIdx.x * blockDim.x + threadIdx.x;           // pair index
  const int NP = NQ_ * (DH_ / 2);
  if (idx >= B_ * NP) return;
  const int b = idx / NP, r = idx % NP, nq = r / (DH_ / 2), i = r % (DH_ / 2);
  const float a  = q_proj[(size_t)nq * DH_ + 2 * i];
  const float bb = q_proj[(size_t)nq * DH_ + 2 * i + 1];
  const float* f = rot_q + ((size_t)b * NQ_ + nq) * DH_;
  const float c0 = __cosf(f[2 * i]),     s0 = __sinf(f[2 * i]);
  const float c1 = __cosf(f[2 * i + 1]), s1 = __sinf(f[2 * i + 1]);
  const size_t base = ((size_t)b * NQ_ + nq) * DH_;
  q_rot[base + 2 * i]     = f2bfu(a * c0 - bb * s0);
  q_rot[base + 2 * i + 1] = f2bfu(bb * c1 + a * s1);
}

// ---------------- layernorm(ctx) -> bf16 ----------------
__global__ void k_ln_ctx(const float* __restrict__ xc,
                         const float* __restrict__ lnw, const float* __restrict__ lnb,
                         unsigned short* __restrict__ xnbf) {
  __shared__ float red[256];
  const int row = blockIdx.x;                 // b*KTOT + kk
  const int tid = threadIdx.x;
  const float xv = xc[(size_t)row * CDIM_ + tid];
  red[tid] = xv; __syncthreads();
  for (int o = 128; o > 0; o >>= 1) { if (tid < o) red[tid] += red[tid + o]; __syncthreads(); }
  const float mu = red[0] * (1.f / CDIM_); __syncthreads();
  const float d = xv - mu;
  red[tid] = d * d; __syncthreads();
  for (int o = 128; o > 0; o >>= 1) { if (tid < o) red[tid] += red[tid + o]; __syncthreads(); }
  const float rstd = rsqrtf(red[0] * (1.f / CDIM_) + 1e-5f);
  xnbf[(size_t)row * CDIM_ + tid] = f2bfu(d * rstd * lnw[tid] + lnb[tid]);
}

// ---------------- KV projection GEMM (WMMA) + rotary + V-transpose epilogue --------
#define KV_RT_PER_WG 8
__global__ __launch_bounds__(256) void k_kv_gemm(const unsigned short* __restrict__ xnbf,
                                                 const unsigned short* __restrict__ wkv,
                                                 const float* __restrict__ rot_c,
                                                 unsigned short* __restrict__ k_rot,
                                                 unsigned short* __restrict__ vT) {
  const int lane = threadIdx.x & 31;
  const int wv = threadIdx.x >> 5;            // 0..7 -> columns [32*wv, 32*wv+32)
  const int col = lane & 15, hi = lane >> 4;

  v16bf wB[2][8];                             // B operand, 2 col-tiles x 8 k-chunks
  #pragma unroll
  for (int ct = 0; ct < 2; ct++) {
    const unsigned short* wr = wkv + (size_t)(32 * wv + 16 * ct + col) * CDIM_;
    #pragma unroll
    for (int c = 0; c < 8; c++) {
      bf16x8 u0 = ld8(wr + 32 * c + 16 * hi);
      bf16x8 u1 = ld8(wr + 32 * c + 16 * hi + 8);
      #pragma unroll
      for (int e = 0; e < 8; e++) { wB[ct][c][e] = u0[e]; wB[ct][c][8 + e] = u1[e]; }
    }
  }

  for (int t = 0; t < KV_RT_PER_WG; t++) {
    const int base_row = (blockIdx.x * KV_RT_PER_WG + t) * 16;
    v8f acc[2];
    #pragma unroll
    for (int ct = 0; ct < 2; ct++)
      #pragma unroll
      for (int r = 0; r < 8; r++) acc[ct][r] = 0.f;

    const unsigned short* arow = xnbf + (size_t)(base_row + col) * CDIM_;
    #pragma unroll
    for (int c = 0; c < 8; c++) {
      v16bf A;
      bf16x8 u0 = ld8(arow + 32 * c + 16 * hi);
      bf16x8 u1 = ld8(arow + 32 * c + 16 * hi + 8);
      #pragma unroll
      for (int e = 0; e < 8; e++) { A[e] = u0[e]; A[8 + e] = u1[e]; }
      acc[0] = __builtin_amdgcn_wmma_f32_16x16x32_bf16(false, A, false, wB[0][c], (short)0, acc[0], false, false);
      acc[1] = __builtin_amdgcn_wmma_f32_16x16x32_bf16(false, A, false, wB[1][c], (short)0, acc[1], false, false);
    }

    // rotary epilogue: pairs are adjacent output cols = adjacent lanes -> shfl_xor(1)
    #pragma unroll
    for (int ct = 0; ct < 2; ct++) {
      const int n = 32 * wv + 16 * ct + col;
      const int dh = n & 127;
      #pragma unroll
      for (int r = 0; r < 8; r++) {
        const int row = base_row + r + 8 * hi;      // b*KTOT + kk
        const float f = rot_c[(size_t)row * DH_ + dh];
        const float val = acc[ct][r];
        const float oth = __shfl_xor(val, 1, 32);
        const float cv = __cosf(f), sv = __sinf(f);
        const float outv = (n & 1) ? (val * cv + oth * sv) : (val * cv - oth * sv);
        if (n < 128) {
          k_rot[(size_t)row * DH_ + n] = f2bfu(outv);
        } else {
          const int b = row >> 13, kk = row & (KTOT_ - 1);
          vT[((size_t)b * DH_ + (n - 128)) * KTOT_ + kk] = f2bfu(outv);
        }
      }
    }
  }
}

// ---------------- additive mask bias ----------------
__global__ void k_maskadd(const int* __restrict__ cmask, float* __restrict__ maskadd) {
  const int idx = blockIdx.x * blockDim.x + threadIdx.x;
  if (idx >= B_ * KTOT_) return;
  const int b = idx / KTOT_, k = idx % KTOT_;
  maskadd[idx] = cmask[b * T_ + (k >> 2)] ? 0.f : -1e30f;
}

// ---------------- flash attention ----------------
// 4 waves/WG share one batch; K/V tiles staged to LDS once per WG (async, double
// buffered); each wave owns 32 queries (2 q-tiles) so K/V registers feed 2 WMMAs each.
__global__ __launch_bounds__(128) void k_attn(const unsigned short* __restrict__ q_rot,
                                              const unsigned short* __restrict__ k_rot,
                                              const unsigned short* __restrict__ vT,
                                              const float* __restrict__ maskadd,
                                              float* __restrict__ att) {
  __shared__ unsigned short kbuf[2][32 * DH_];   // 2 x 8 KB   (keys x dh, row-major)
  __shared__ unsigned short vbuf[2][DH_ * 32];   // 2 x 8 KB   (dh x keys, transposed)

  const int lane = threadIdx.x & 31;
  const int tid  = threadIdx.x;
  const int wid  = blockIdx.x * (blockDim.x >> 5) + (threadIdx.x >> 5);
  const int b = wid >> 5;                     // 32 q-parts per batch; 4/WG share batch
  const int qbase = (wid & 31) * 32;
  const int col = lane & 15;
  const int hi  = lane >> 4;

  const unsigned short* kb = k_rot + (size_t)b * KTOT_ * DH_;
  const unsigned short* vb = vT    + (size_t)b * DH_ * KTOT_;
  const float* mb = maskadd + (size_t)b * KTOT_;

  // Q in B-layout, 2 q-tiles, fixed for the wave
  v16bf qB[2][4];
  #pragma unroll
  for (int qt = 0; qt < 2; qt++) {
    const unsigned short* qrow = q_rot + ((size_t)b * NQ_ + qbase + 16 * qt + col) * DH_;
    #pragma unroll
    for (int c = 0; c < 4; c++) {
      bf16x8 u0 = ld8(qrow + 32 * c + 16 * hi);
      bf16x8 u1 = ld8(qrow + 32 * c + 16 * hi + 8);
      #pragma unroll
      for (int e = 0; e < 8; e++) { qB[qt][c][e] = u0[e]; qB[qt][c][8 + e] = u1[e]; }
    }
  }

  v8f O[2][8];
  #pragma unroll
  for (int qt = 0; qt < 2; qt++)
    #pragma unroll
    for (int dch = 0; dch < 8; dch++)
      #pragma unroll
      for (int r = 0; r < 8; r++) O[qt][dch][r] = 0.f;

  float m_run[2] = { -3.0e38f, -3.0e38f };
  float l_run[2] = { 0.f, 0.f };
  const float scale = 0.08838834764831845f;   // 1/sqrt(128)

  // stage one 32-key K/V tile: 1024 x b128 transfers spread over 128 threads
  auto issue_tile = [&](int k0, int bufi) {
    #pragma unroll
    for (int i = 0; i < 4; i++) {
      const int id = i * 128 + tid;
      const int rk = id >> 4, ok = (id & 15) * 8;          // K: 32 rows x 128 dh
      cp16_to_lds(kb + (size_t)(k0 + rk) * DH_ + ok, &kbuf[bufi][rk * DH_ + ok]);
    }
    #pragma unroll
    for (int i = 0; i < 4; i++) {
      const int id = i * 128 + tid;
      const int rv = id >> 2, ov = (id & 3) * 8;           // V^T: 128 rows x 32 keys
      cp16_to_lds(vb + (size_t)rv * KTOT_ + k0 + ov, &vbuf[bufi][rv * 32 + ov]);
    }
  };

  issue_tile(0, 0);
  const int NT = KTOT_ / 32;
  for (int it = 0; it < NT; it++) {
    const int k0 = it * 32;
    const int cur = it & 1;
    const bool more = (it + 1) < NT;
    if (more) issue_tile(k0 + 32, cur ^ 1);
#if ASYNC_LDS
    if (more) __builtin_amdgcn_s_wait_asynccnt(8);   // older 8 (this tile) done
    else      __builtin_amdgcn_s_wait_asynccnt(0);
#endif
    __syncthreads();

    // K tile in A-layout from LDS, kept for both q-tiles
    v16bf kA[2][4];
    #pragma unroll
    for (int j = 0; j < 2; j++) {
      const unsigned short* krow = &kbuf[cur][(16 * j + col) * DH_];
      #pragma unroll
      for (int c = 0; c < 4; c++) {
        bf16x8 u0 = ld8(krow + 32 * c + 16 * hi);
        bf16x8 u1 = ld8(krow + 32 * c + 16 * hi + 8);
        #pragma unroll
        for (int e = 0; e < 8; e++) { kA[j][c][e] = u0[e]; kA[j][c][8 + e] = u1[e]; }
      }
    }

    v16bf pA[2];
    const float* mp = mb + k0 + 8 * hi;
    #pragma unroll
    for (int qt = 0; qt < 2; qt++) {
      // S^T = K_tile x Q^T
      v8f sT0, sT1;
      #pragma unroll
      for (int r = 0; r < 8; r++) { sT0[r] = 0.f; sT1[r] = 0.f; }
      #pragma unroll
      for (int c = 0; c < 4; c++) {
        sT0 = __builtin_amdgcn_wmma_f32_16x16x32_bf16(false, kA[0][c], false, qB[qt][c], (short)0, sT0, false, false);
        sT1 = __builtin_amdgcn_wmma_f32_16x16x32_bf16(false, kA[1][c], false, qB[qt][c], (short)0, sT1, false, false);
      }
      // masked logits + online softmax (key = k0 + 16j + r + 8*hi, query = col)
      float p0[8], p1[8];
      float tmax = -3.0e38f;
      #pragma unroll
      for (int r = 0; r < 8; r++) {
        float l0 = sT0[r] * scale + mp[r];
        float l1 = sT1[r] * scale + mp[16 + r];
        p0[r] = l0; p1[r] = l1;
        tmax = fmaxf(tmax, fmaxf(l0, l1));
      }
      tmax = fmaxf(tmax, __shfl_xor(tmax, 16, 32));
      const float m_new = fmaxf(m_run[qt], tmax);
      const float alpha = __expf(m_run[qt] - m_new);
      float lsum = 0.f;
      #pragma unroll
      for (int r = 0; r < 8; r++) {
        p0[r] = __expf(p0[r] - m_new);
        p1[r] = __expf(p1[r] - m_new);
        lsum += p0[r] + p1[r];
      }
      lsum += __shfl_xor(lsum, 16, 32);
      l_run[qt] = l_run[qt] * alpha + lsum;
      m_run[qt] = m_new;

      float ar[8];
      #pragma unroll
      for (int r = 0; r < 8; r++) ar[r] = __shfl(alpha, r + 8 * hi, 32);
      #pragma unroll
      for (int dch = 0; dch < 8; dch++)
        #pragma unroll
        for (int r = 0; r < 8; r++) O[qt][dch][r] *= ar[r];

      #pragma unroll
      for (int e = 0; e < 8; e++) { pA[qt][e] = f2bf(p0[e]); pA[qt][8 + e] = f2bf(p1[e]); }
    }

    // O += P x V ; V tile read once from LDS, used by both q-tiles
    #pragma unroll
    for (int dch = 0; dch < 8; dch++) {
      const unsigned short* vp = &vbuf[cur][(16 * dch + col) * 32 + 8 * hi];
      v16bf vB;
      bf16x8 u0 = ld8(vp);
      bf16x8 u1 = ld8(vp + 16);
      #pragma unroll
      for (int e = 0; e < 8; e++) { vB[e] = u0[e]; vB[8 + e] = u1[e]; }
      O[0][dch] = __builtin_amdgcn_wmma_f32_16x16x32_bf16(false, pA[0], false, vB, (short)0, O[0][dch], false, false);
      O[1][dch] = __builtin_amdgcn_wmma_f32_16x16x32_bf16(false, pA[1], false, vB, (short)0, O[1][dch], false, false);
    }
    __syncthreads();                      // tile fully consumed before overwrite
  }

  // normalize and store
  #pragma unroll
  for (int qt = 0; qt < 2; qt++) {
    float lr[8];
    #pragma unroll
    for (int r = 0; r < 8; r++) {
      float lq = __shfl(l_run[qt], r + 8 * hi, 32);
      lr[r] = (lq > 0.f) ? (1.f / lq) : 0.f;
    }
    #pragma unroll
    for (int dch = 0; dch < 8; dch++)
      #pragma unroll
      for (int r = 0; r < 8; r++) {
        const int q  = qbase + 16 * qt + r + 8 * hi;
        const int dh = 16 * dch + col;
        att[((size_t)b * NQ_ + q) * DH_ + dh] = O[qt][dch][r] * lr[r];
      }
  }
}

// ---------------- inverse rotary on attention output -> bf16 ----------------
__global__ void k_att_rot(const float* __restrict__ att, const float* __restrict__ rot_q,
                          unsigned short* __restrict__ att_rot) {
  const int idx = blockIdx.x * blockDim.x + threadIdx.x;   // pair index
  if (idx >= B_ * NQ_ * (DH_ / 2)) return;
  const int row = idx / (DH_ / 2), i = idx % (DH_ / 2);
  const float a  = att[(size_t)row * DH_ + 2 * i];
  const float bb = att[(size_t)row * DH_ + 2 * i + 1];
  const float* f = rot_q + (size_t)row * DH_;
  const float c0 = __cosf(f[2 * i]),     s0 = __sinf(f[2 * i]);
  const float c1 = __cosf(f[2 * i + 1]), s1 = __sinf(f[2 * i + 1]);
  att_rot[(size_t)row * DH_ + 2 * i]     = f2bfu(a * c0 + bb * s0);   // apply_rotary(-f, x)
  att_rot[(size_t)row * DH_ + 2 * i + 1] = f2bfu(bb * c1 - a * s1);
}

// ---------------- O projection GEMM (WMMA): out = att_rot @ w_o^T + b_o ----------------
#define O_RT_PER_WG 4
__global__ __launch_bounds__(256) void k_o_gemm(const unsigned short* __restrict__ arot,
                                                const unsigned short* __restrict__ wo,
                                                const float* __restrict__ b_o,
                                                float* __restrict__ out) {
  const int lane = threadIdx.x & 31;
  const int wv = threadIdx.x >> 5;            // 0..7 -> columns [64*wv, 64*wv+64)
  const int col = lane & 15, hi = lane >> 4;

  v16bf wB[4][4];                             // 4 col-tiles x 4 k-chunks (K=128)
  float bn[4];
  #pragma unroll
  for (int ct = 0; ct < 4; ct++) {
    const int n = 64 * wv + 16 * ct + col;
    bn[ct] = b_o[n];
    const unsigned short* wr = wo + (size_t)n * INNER_;
    #pragma unroll
    for (int c = 0; c < 4; c++) {
      bf16x8 u0 = ld8(wr + 32 * c + 16 * hi);
      bf16x8 u1 = ld8(wr + 32 * c + 16 * hi + 8);
      #pragma unroll
      for (int e = 0; e < 8; e++) { wB[ct][c][e] = u0[e]; wB[ct][c][8 + e] = u1[e]; }
    }
  }

  for (int t = 0; t < O_RT_PER_WG; t++) {
    const int base_row = (blockIdx.x * O_RT_PER_WG + t) * 16;
    v8f acc[4];
    #pragma unroll
    for (int ct = 0; ct < 4; ct++)
      #pragma unroll
      for (int r = 0; r < 8; r++) acc[ct][r] = 0.f;

    const unsigned short* arow = arot + (size_t)(base_row + col) * INNER_;
    #pragma unroll
    for (int c = 0; c < 4; c++) {
      v16bf A;
      bf16x8 u0 = ld8(arow + 32 * c + 16 * hi);
      bf16x8 u1 = ld8(arow + 32 * c + 16 * hi + 8);
      #pragma unroll
      for (int e = 0; e < 8; e++) { A[e] = u0[e]; A[8 + e] = u1[e]; }
      #pragma unroll
      for (int ct = 0; ct < 4; ct++)
        acc[ct] = __builtin_amdgcn_wmma_f32_16x16x32_bf16(false, A, false, wB[ct][c], (short)0, acc[ct], false, false);
    }

    #pragma unroll
    for (int ct = 0; ct < 4; ct++) {
      const int n = 64 * wv + 16 * ct + col;
      #pragma unroll
      for (int r = 0; r < 8; r++) {
        const int row = base_row + r + 8 * hi;
        out[(size_t)row * DIM_ + n] = acc[ct][r] + bn[ct];
      }
    }
  }
}

extern "C" void kernel_launch(void* const* d_in, const int* in_sizes, int n_in,
                              void* d_out, int out_size, void* d_ws, size_t ws_size,
                              hipStream_t stream) {
  const float* x_query = (const float*)d_in[0];
  const float* x_ctx   = (const float*)d_in[1];
  const float* rot_q   = (const float*)d_in[2];
  const float* rot_c   = (const float*)d_in[3];
  const int*   cmask   = (const int*)d_in[4];
  const float* ln_q_w  = (const float*)d_in[5];
  const float* ln_q_b  = (const float*)d_in[6];
  const float* ln_c_w  = (const float*)d_in[7];
  const float* ln_c_b  = (const float*)d_in[8];
  const float* w_q     = (const float*)d_in[9];
  const float* w_kv    = (const float*)d_in[10];
  const float* w_o     = (const float*)d_in[11];
  const float* b_o     = (const float*)d_in[12];
  float* out = (float*)d_out;
  (void)in_sizes; (void)n_in; (void)out_size; (void)ws_size;

  char* ws = (char*)d_ws;
  size_t off = 0;
  auto carve = [&](size_t bytes) -> char* {
    char* p = ws + off; off += (bytes + 255) & ~(size_t)255; return p;
  };
  float*          q_proj  = (float*)carve(sizeof(float) * NQ_ * INNER_);
  unsigned short* q_rot   = (unsigned short*)carve(sizeof(short) * (size_t)B_*NQ_*DH_);
  unsigned short* k_rotb  = (unsigned short*)carve(sizeof(short) * (size_t)B_*KTOT_*DH_);
  unsigned short* vT      = (unsigned short*)carve(sizeof(short) * (size_t)B_*DH_*KTOT_);
  float*          maskadd = (float*)carve(sizeof(float) * (size_t)B_*KTOT_);
  unsigned short* wkv_bf  = (unsigned short*)carve(sizeof(short) * (size_t)(2*INNER_)*CDIM_);
  unsigned short* wo_bf   = (unsigned short*)carve(sizeof(short) * (size_t)DIM_*INNER_);
  unsigned short* xnbf    = (unsigned short*)carve(sizeof(short) * (size_t)B_*KTOT_*CDIM_);
  float*          att     = (float*)carve(sizeof(float) * (size_t)B_*NQ_*DH_);
  unsigned short* att_rot = (unsigned short*)carve(sizeof(short) * (size_t)B_*NQ_*DH_);

  // weight conversions
  k_convert<<<(2*INNER_*CDIM_ + 255) / 256, 256, 0, stream>>>(w_kv, wkv_bf, 2*INNER_*CDIM_);
  k_convert<<<(DIM_*INNER_ + 255) / 256, 256, 0, stream>>>(w_o, wo_bf, DIM_*INNER_);

  // Q path
  k_ln_q_proj<<<NQ_, 256, 0, stream>>>(x_query, ln_q_w, ln_q_b, w_q, q_proj);
  const int npair = B_ * NQ_ * (DH_ / 2);
  k_rot_q<<<(npair + 255) / 256, 256, 0, stream>>>(q_proj, rot_q, q_rot);

  // KV path: LN -> WMMA GEMM (+rotary, V transposed)
  k_ln_ctx<<<B_ * KTOT_, 256, 0, stream>>>(x_ctx, ln_c_w, ln_c_b, xnbf);
  k_kv_gemm<<<(B_ * KTOT_ / 16) / KV_RT_PER_WG, 256, 0, stream>>>(xnbf, wkv_bf, rot_c, k_rotb, vT);

  // mask + flash attention (LDS-staged, double-buffered K/V)
  k_maskadd<<<(B_ * KTOT_ + 255) / 256, 256, 0, stream>>>(cmask, maskadd);
  k_attn<<<(B_ * (NQ_ / 32)) / 4, 128, 0, stream>>>(q_rot, k_rotb, vT, maskadd, att);

  // inverse rotary + O projection (WMMA)
  k_att_rot<<<(npair + 255) / 256, 256, 0, stream>>>(att, rot_q, att_rot);
  k_o_gemm<<<(B_ * NQ_ / 16) / O_RT_PER_WG, 256, 0, stream>>>(att_rot, wo_bf, b_o, out);
}
